// PointNet2DecoderPyG_13237089206889
// MI455X (gfx1250) — matrix operand order repaired and checked by reading, not compiled, not run on hardware
//
#include <hip/hip_runtime.h>
#include <hip/hip_bf16.h>
#include <stdint.h>

// ---------------------------------------------------------------------------
// PointNet++ FP decoder on gfx1250.
// GEMMs use v_wmma_f32_16x16x32_bf16 (fp32 accum). bf16 A/B is acceptable
// because every linear is followed by BatchNorm which renormalizes per channel.
// Total ~6.4 GFLOP / tens of MB traffic -> latency-bound; keep kernels simple,
// keep matrix math on the WMMA pipe.
// ---------------------------------------------------------------------------

typedef __bf16 bf16_t;
typedef __attribute__((ext_vector_type(16))) __bf16 v16bf;
typedef __attribute__((ext_vector_type(8)))  float  v8f;

#define BN_EPS 1e-5f

// ------------------------- kNN(3) interpolation ----------------------------
__global__ void knn3_interp(const float* __restrict__ p_up,
                            const float* __restrict__ p_dn,
                            const float* __restrict__ x_dn,
                            float* __restrict__ out,
                            int n_up, int n_dn, int C, int B)
{
  int i = blockIdx.x * blockDim.x + threadIdx.x;
  int total = B * n_up;
  if (i >= total) return;
  int b = i / n_up;
  float px = p_up[3 * (size_t)i + 0];
  float py = p_up[3 * (size_t)i + 1];
  float pz = p_up[3 * (size_t)i + 2];
  const float* pd = p_dn + (size_t)b * n_dn * 3;

  float d0 = 3.4e38f, d1 = 3.4e38f, d2 = 3.4e38f;
  int j0 = 0, j1 = 0, j2 = 0;
  for (int j = 0; j < n_dn; ++j) {
    float dx = px - pd[3 * j + 0];
    float dy = py - pd[3 * j + 1];
    float dz = pz - pd[3 * j + 2];
    float d = dx * dx + dy * dy + dz * dz;
    if (d < d0)      { d2 = d1; j2 = j1; d1 = d0; j1 = j0; d0 = d; j0 = j; }
    else if (d < d1) { d2 = d1; j2 = j1; d1 = d;  j1 = j; }
    else if (d < d2) { d2 = d;  j2 = j; }
  }
  float w0 = 1.f / fmaxf(d0, 1e-16f);
  float w1 = 1.f / fmaxf(d1, 1e-16f);
  float w2 = 1.f / fmaxf(d2, 1e-16f);
  float ws = 1.f / (w0 + w1 + w2);

  const float* xb = x_dn + (size_t)b * n_dn * C;
  const float* r0 = xb + (size_t)j0 * C;
  const float* r1 = xb + (size_t)j1 * C;
  const float* r2 = xb + (size_t)j2 * C;
  float* o = out + (size_t)i * C;
  for (int c = 0; c < C; ++c)
    o[c] = (w0 * r0[c] + w1 * r1[c] + w2 * r2[c]) * ws;
}

// --------------------- concat two fp32 mats -> bf16 ------------------------
__global__ void pack_cat_bf16(const float* __restrict__ xa,
                              const float* __restrict__ xb,
                              bf16_t* __restrict__ out,
                              int M, int Ca, int Cb)
{
  int C = Ca + Cb;
  size_t idx = (size_t)blockIdx.x * blockDim.x + threadIdx.x;
  size_t total = (size_t)M * C;
  if (idx >= total) return;
  int c = (int)(idx % C);
  int r = (int)(idx / C);
  float v = (c < Ca) ? xa[(size_t)r * Ca + c] : xb[(size_t)r * Cb + (c - Ca)];
  out[idx] = (bf16_t)v;
}

// --------------- transpose weight K x N fp32 -> N x K bf16 -----------------
__global__ void pack_wt_bf16(const float* __restrict__ W,
                             bf16_t* __restrict__ Wt, int K, int N)
{
  size_t idx = (size_t)blockIdx.x * blockDim.x + threadIdx.x;
  size_t total = (size_t)K * N;
  if (idx >= total) return;
  int k = (int)(idx / N);
  int n = (int)(idx % N);
  Wt[(size_t)n * K + k] = (bf16_t)W[idx];
}

// --------------------------- WMMA bf16 GEMM --------------------------------
// A: M x K row-major bf16, Wt: N x K row-major bf16 (i.e., W transposed),
// Y = A @ W + bias, fp32. One wave computes one 16x16 tile of Y; 4 waves per
// block cover 64 rows. A-operand lane layout (16-bit A 16x32): lanes 0-15 hold
// M=lane, K={0..7,16..23}; lanes 16-31 hold M=lane-16, K={8..15,24..31}.
// B-operand (32x16): lanes 0-15 hold N=lane, K={0..7,16..23}; lanes 16-31 K={8..15,24..31}.
// Both map to two contiguous 16-byte chunks per lane -> b128 loads.
union ABits { v16bf v; uint4 q[2]; };

__global__ void gemm_bf16_wmma(const bf16_t* __restrict__ A,
                               const bf16_t* __restrict__ Wt,
                               const float* __restrict__ bias,
                               float* __restrict__ Y,
                               int M, int N, int K)
{
  int lane = threadIdx.x & 31;
  int wave = threadIdx.x >> 5;
  int half = lane >> 4;
  int l15  = lane & 15;
  int bm = (blockIdx.y * 4 + wave) * 16;
  int bn = blockIdx.x * 16;

  v8f acc = {0.f, 0.f, 0.f, 0.f, 0.f, 0.f, 0.f, 0.f};
  const bf16_t* Arow = A  + (size_t)(bm + l15) * K;
  const bf16_t* Brow = Wt + (size_t)(bn + l15) * K;

  for (int ks = 0; ks < K; ks += 32) {
    ABits a, b;
    const uint4* a4 = (const uint4*)(Arow + ks);
    const uint4* b4 = (const uint4*)(Brow + ks);
    // A lane data: dwords [4*half .. 4*half+3] and [8+4*half .. 11+4*half]
    a.q[0] = a4[half];
    a.q[1] = a4[2 + half];
    // B lane data: dwords [4*half .. 4*half+3] and [8+4*half .. 11+4*half]
    b.q[0] = b4[half];
    b.q[1] = b4[2 + half];
    acc = __builtin_amdgcn_wmma_f32_16x16x32_bf16(
        /*neg_a=*/false, a.v, /*neg_b=*/false, b.v,
        /*c_mod=*/(short)0, acc, /*reuse_a=*/false, /*reuse_b=*/false);
  }

  int n = bn + l15;
  float bv = bias[n];
#pragma unroll
  for (int r = 0; r < 8; ++r) {
    int m = bm + r + 8 * half;   // C/D layout: lanes 0-15 -> M=r, 16-31 -> M=r+8
    Y[(size_t)m * N + n] = acc[r] + bv;
  }
}

// ----------------------- BatchNorm batch statistics ------------------------
__global__ void bn_stats(const float* __restrict__ Y,
                         float* __restrict__ mean, float* __restrict__ invstd,
                         int M, int N)
{
  __shared__ float s1[256];
  __shared__ float s2[256];
  int c = blockIdx.x;
  float sum = 0.f, sq = 0.f;
  for (int r = threadIdx.x; r < M; r += blockDim.x) {
    float v = Y[(size_t)r * N + c];
    sum += v; sq += v * v;
  }
  s1[threadIdx.x] = sum;
  s2[threadIdx.x] = sq;
  __syncthreads();
  for (int s = blockDim.x >> 1; s > 0; s >>= 1) {
    if ((int)threadIdx.x < s) {
      s1[threadIdx.x] += s1[threadIdx.x + s];
      s2[threadIdx.x] += s2[threadIdx.x + s];
    }
    __syncthreads();
  }
  if (threadIdx.x == 0) {
    float mu  = s1[0] / (float)M;
    float var = s2[0] / (float)M - mu * mu;
    mean[c]   = mu;
    invstd[c] = rsqrtf(fmaxf(var, 0.f) + BN_EPS);
  }
}

// --------------------------- BN apply + ReLU -------------------------------
__global__ void bn_relu(const float* __restrict__ Y,
                        const float* __restrict__ mean,
                        const float* __restrict__ invstd,
                        const float* __restrict__ gamma,
                        const float* __restrict__ beta,
                        float* __restrict__ X, int M, int N)
{
  size_t idx = (size_t)blockIdx.x * blockDim.x + threadIdx.x;
  size_t total = (size_t)M * N;
  if (idx >= total) return;
  int c = (int)(idx % N);
  float v = (Y[idx] - mean[c]) * invstd[c] * gamma[c] + beta[c];
  X[idx] = fmaxf(v, 0.f);
}

// ---------------------------------------------------------------------------
// Host orchestration
// ---------------------------------------------------------------------------
static inline unsigned cdiv(size_t a, size_t b) { return (unsigned)((a + b - 1) / b); }

// workspace layout (bytes)
static constexpr size_t OFF_WT     = 0;                       // 688128 B bf16 weights
static constexpr size_t OFF_STATS  = 720896;                  // mean(512f)+invstd(512f)
static constexpr size_t OFF_XPREV  = 786432;                  // 8 MiB   coarse activations
static constexpr size_t OFF_INTERP = 9437184;                 // 32 MiB  interp / xmid
static constexpr size_t OFF_CAT    = 42991616;                // 24 MiB  bf16 concat
static constexpr size_t OFF_Y      = 68157440;                // 16 MiB  pre-BN
// total ~81 MiB

extern "C" void kernel_launch(void* const* d_in, const int* in_sizes, int n_in,
                              void* d_out, int out_size, void* d_ws, size_t ws_size,
                              hipStream_t stream)
{
  const int B = 8;
  const float* p0 = (const float*)d_in[0];
  const float* x0 = (const float*)d_in[1];
  const float* p1 = (const float*)d_in[2];
  const float* x1 = (const float*)d_in[3];
  const float* p2 = (const float*)d_in[4];
  const float* x2 = (const float*)d_in[5];
  const float* p3 = (const float*)d_in[6];
  const float* x3 = (const float*)d_in[7];
  // params pytree leaves, dict keys sorted: W, b, beta, gamma per layer;
  // mlp0 layers at d_in[8..15], mlp1 at [16..23], mlp2 at [24..31].

  char*   ws     = (char*)d_ws;
  bf16_t* wt     = (bf16_t*)(ws + OFF_WT);
  float*  meanp  = (float*)(ws + OFF_STATS);
  float*  istdp  = meanp + 512;
  float*  xprev  = (float*)(ws + OFF_XPREV);
  float*  interp = (float*)(ws + OFF_INTERP);
  bf16_t* cat    = (bf16_t*)(ws + OFF_CAT);
  float*  ybuf   = (float*)(ws + OFF_Y);

  // bf16 transposed weight slabs (element offsets)
  bf16_t* wt2_0 = wt + 0;        // 768x256
  bf16_t* wt2_1 = wt + 196608;   // 256x256
  bf16_t* wt1_0 = wt + 262144;   // 384x128
  bf16_t* wt1_1 = wt + 311296;   // 128x128
  bf16_t* wt0_0 = wt + 327680;   // 192x64
  bf16_t* wt0_1 = wt + 339968;   // 64x64

  struct WtJob { int in_idx; bf16_t* dst; int K, N; };
  const WtJob wjobs[6] = {
    {24, wt2_0, 768, 256}, {28, wt2_1, 256, 256},
    {16, wt1_0, 384, 128}, {20, wt1_1, 128, 128},
    { 8, wt0_0, 192,  64}, {12, wt0_1,  64,  64},
  };
  for (int j = 0; j < 6; ++j) {
    size_t tot = (size_t)wjobs[j].K * wjobs[j].N;
    pack_wt_bf16<<<cdiv(tot, 256), 256, 0, stream>>>(
        (const float*)d_in[wjobs[j].in_idx], wjobs[j].dst, wjobs[j].K, wjobs[j].N);
  }

  auto run_layer = [&](const bf16_t* Ain, int M, int K, int N, const bf16_t* Wt_l,
                       const float* bias, const float* beta, const float* gamma,
                       float* Xout) {
    dim3 grid(N / 16, M / 64);
    gemm_bf16_wmma<<<grid, 128, 0, stream>>>(Ain, Wt_l, bias, ybuf, M, N, K);
    bn_stats<<<N, 256, 0, stream>>>(ybuf, meanp, istdp, M, N);
    size_t tot = (size_t)M * N;
    bn_relu<<<cdiv(tot, 256), 256, 0, stream>>>(ybuf, meanp, istdp, gamma, beta,
                                                Xout, M, N);
  };

  auto run_level = [&](const float* pup, const float* pdn, const float* xdn,
                       const float* xup, int nup, int ndn, int Cdn, int Cup,
                       int pbase, const bf16_t* wtA, const bf16_t* wtB,
                       int N1, int N2, float* xout) {
    int Mtot = B * nup;
    knn3_interp<<<cdiv((size_t)Mtot, 256), 256, 0, stream>>>(
        pup, pdn, xdn, interp, nup, ndn, Cdn, B);
    size_t tot1 = (size_t)Mtot * (Cup + Cdn);
    pack_cat_bf16<<<cdiv(tot1, 256), 256, 0, stream>>>(xup, interp, cat, Mtot, Cup, Cdn);
    run_layer(cat, Mtot, Cup + Cdn, N1, wtA,
              (const float*)d_in[pbase + 1], (const float*)d_in[pbase + 2],
              (const float*)d_in[pbase + 3], interp /*xmid reuses interp arena*/);
    size_t tot2 = (size_t)Mtot * N1;
    pack_cat_bf16<<<cdiv(tot2, 256), 256, 0, stream>>>(interp, interp, cat, Mtot, N1, 0);
    run_layer(cat, Mtot, N1, N2, wtB,
              (const float*)d_in[pbase + 5], (const float*)d_in[pbase + 6],
              (const float*)d_in[pbase + 7], xout);
  };

  // coarsest-first, mirroring the reference reverse loop
  // level 2: x2n = MLP([x2, knn(x3: p3 -> p2)])   768 -> 256 -> 256
  run_level(p2, p3, x3, x2, 512, 128, 512, 256, 24, wt2_0, wt2_1, 256, 256, xprev);
  // level 1: x1n = MLP([x1, knn(x2n: p2 -> p1)])  384 -> 128 -> 128
  run_level(p1, p2, xprev, x1, 2048, 512, 256, 128, 16, wt1_0, wt1_1, 128, 128, xprev);
  // level 0: out = MLP([x0, knn(x1n: p1 -> p0)])  192 -> 64 -> 64
  run_level(p0, p1, xprev, x0, 8192, 2048, 128, 64, 8, wt0_0, wt0_1, 64, 64,
            (float*)d_out);

  (void)in_sizes; (void)n_in; (void)out_size; (void)ws_size;
}
